// NeuralHawkes_82600811036853
// MI455X (gfx1250) — matrix operand order repaired and checked by reading, not compiled, not run on hardware
//
#include <hip/hip_runtime.h>
#include <hip/hip_bf16.h>

#define EPSF 2.220446049250313e-16f

typedef __attribute__((ext_vector_type(16))) __bf16 v16bf;
typedef __attribute__((ext_vector_type(16))) unsigned short u16x16;
typedef __attribute__((ext_vector_type(8)))  unsigned short u16x8;
typedef __attribute__((ext_vector_type(8)))  float v8f;
typedef __attribute__((ext_vector_type(4)))  unsigned int u32x4;
typedef __attribute__((ext_vector_type(8)))  int i32x8;
typedef __attribute__((ext_vector_type(4)))  int i32x4;

#if __has_builtin(__builtin_amdgcn_tensor_load_to_lds) && \
    __has_builtin(__builtin_amdgcn_s_wait_tensorcnt)
#define HAVE_TDM 1
#warning "CDNA5 probe: TDM tensor_load_to_lds path ENABLED"
#else
#define HAVE_TDM 0
#warning "CDNA5 probe: TDM tensor_load_to_lds path DISABLED (builtin missing)"
#endif

// ---------- helpers ----------
__device__ __forceinline__ unsigned short f2bf(float x) {
    union { float f; unsigned u; } v; v.f = x;
    unsigned r = v.u + 0x7FFFu + ((v.u >> 16) & 1u);
    return (unsigned short)(r >> 16);
}
__device__ __forceinline__ float bf2f(unsigned short h) {
    union { unsigned u; float f; } v; v.u = ((unsigned)h) << 16; return v.f;
}
__device__ __forceinline__ float sigm(float x) { return 1.0f / (1.0f + __expf(-x)); }
__device__ __forceinline__ float softplus(float x) {
    return fmaxf(x, 0.0f) + log1pf(__expf(-fabsf(x)));
}

__device__ __forceinline__ v8f wmma_bf16(v16bf a, v16bf b, v8f c) {
    return __builtin_amdgcn_wmma_f32_16x16x32_bf16(false, a, false, b, (short)0, c,
                                                   false, false);
}

// A-frag (16x32 bf16): lane L<16 -> row M=L, Ks kb..kb+15;
// lane L>=16 -> row M=L-16, Ks kb+16..kb+31. `row` points at the row start.
template <typename PTR>
__device__ __forceinline__ v16bf load_a_frag(PTR row, int kb, int lane) {
    const int kbase = kb + ((lane < 16) ? 0 : 16);
    u16x16 u;
#pragma unroll
    for (int e = 0; e < 8; e++) {
        u[e]     = row[kbase + e];
        u[8 + e] = row[kbase + 8 + e];
    }
    return __builtin_bit_cast(v16bf, u);
}

// B-frag (32x16 bf16, K x N) from TRANSPOSED weights Wt[n][k] (row stride 512):
// lane L<16 -> K=k0..k0+15, lane>=16 -> K=k0+16..k0+31, column N = lane&15.
// Per lane: 16 contiguous bf16 = 32 aligned bytes -> 2x global_load_b128.
__device__ __forceinline__ v16bf load_b_frag_t(const unsigned short* Wt, int k0, int n,
                                               int lane) {
    const unsigned short* p = Wt + (size_t)n * 512 + k0 + ((lane < 16) ? 0 : 16);
    u16x16 u;
#pragma unroll
    for (int e = 0; e < 16; e++) u[e] = p[e];
    return __builtin_bit_cast(v16bf, u);
}

// TDM: async-DMA one step's pre slab (57344 bf16 = 114688 B) global -> LDS.
// D# per CDNA5 ISA: 2D tensor, 1-row tile.
__device__ __forceinline__ void tdm_load_pre(const unsigned short* gsrc,
                                             unsigned lds_off) {
#if HAVE_TDM
    const unsigned long long ga = (unsigned long long)(size_t)gsrc;
    u32x4 g0;
    g0[0] = 1u;                                        // count=1, user D#
    g0[1] = lds_off;                                   // lds_addr (bytes)
    g0[2] = (unsigned)(ga & 0xFFFFFFFFu);              // global_addr[31:0]
    g0[3] = (unsigned)((ga >> 32) & 0x1FFFFFFu) | (2u << 30);  // addr[56:32]|type=2
    // data_size=1(2B); tensor_dim0=tile_dim0=57344; tensor_dim1=tile_dim1=1;
    // tensor_dim0_stride=57344
    const i32x8 g1 = {0x00010000, (int)0xE0000000, 0x00010000, (int)0xE0000000,
                      1, 57344, 0, 0};
    const i32x4 z4 = {0, 0, 0, 0};
#if defined(__clang_major__) && (__clang_major__ >= 23)
    const i32x8 z8 = {0, 0, 0, 0, 0, 0, 0, 0};
    __builtin_amdgcn_tensor_load_to_lds(g0, g1, z4, z4, z8, 0);
#else
    __builtin_amdgcn_tensor_load_to_lds(g0, g1, z4, z4, 0);
#endif
#else
    (void)gsrc; (void)lds_off;
#endif
}

// ---------- kernel 1: convert W (transposed) and Emb to bf16 ----------
__global__ void convert_bf16(const float* __restrict__ W, const float* __restrict__ Emb,
                             unsigned short* __restrict__ Wt,
                             unsigned short* __restrict__ Embbf) {
    int i = blockIdx.x * blockDim.x + threadIdx.x;
    if (i < 512 * 1792) {
        const int k = i / 1792, n = i % 1792;
        Wt[(size_t)n * 512 + k] = f2bf(W[i]);
    }
    if (i < 23 * 256) Embbf[i] = f2bf(Emb[i]);
}

// ---------- kernel 2: pre[t*32+b][n] = bf16(Emb[event] @ W_top + bias) ----------
// M = 16352, N = 1792, K = 256. One wave per 16x64 strip (1 Mtile x 4 Ntiles).
__global__ void pre_gemm(const int* __restrict__ event,
                         const unsigned short* __restrict__ Embbf,
                         const unsigned short* __restrict__ Wt,
                         const float* __restrict__ bias,
                         unsigned short* __restrict__ pre_bf) {
    const int wave = (blockIdx.x * blockDim.x + threadIdx.x) >> 5;
    const int lane = threadIdx.x & 31;
    const int NG = 28;                       // 112 Ntiles / 4
    const int mt = wave / NG, ng = wave % NG;
    if (mt >= 1022) return;                  // 16352 / 16

    const int row = mt * 16 + (lane & 15);   // row = t*32 + b
    const int t = row >> 5, b = row & 31;
    const int ev = event[b * 512 + t];       // event[b][0][t]
    const unsigned short* arow = Embbf + (size_t)ev * 256;

    v16bf a[8];
#pragma unroll
    for (int kf = 0; kf < 8; kf++) a[kf] = load_a_frag(arow, kf * 32, lane);

    const int mbase = mt * 16 + ((lane < 16) ? 0 : 8);
#pragma unroll
    for (int nt4 = 0; nt4 < 4; nt4++) {
        const int n = (ng * 4 + nt4) * 16 + (lane & 15);
        v8f acc = {};
#pragma unroll
        for (int kf = 0; kf < 8; kf++) {
            v16bf bm = load_b_frag_t(Wt, kf * 32, n, lane);  // K rows 0..255
            acc = wmma_bf16(a[kf], bm, acc);
        }
        const float bn = bias[n];
#pragma unroll
        for (int r = 0; r < 8; r++)
            pre_bf[(size_t)(mbase + r) * 1792 + n] = f2bf(acc[r] + bn);
    }
}

// ---------- kernel 3: persistent recurrent scan (single WGP, 32 waves) ----------
// LDS: z bf16 (112KB) + pre stage bf16 (112KB) + h bf16 (16KB) = 240KB of 320KB.
// c/cbar in registers. TDM streams pre_t into LDS overlapped with the WMMA GEMM.
// Gate phase: thread owns (b = tid/32, 8 consecutive j) -> all LDS traffic is b128.
__global__ void __launch_bounds__(1024, 1)
recurrent_scan(const float* __restrict__ dtime, const unsigned short* __restrict__ Wt,
               const unsigned short* __restrict__ pre_bf,
               unsigned short* __restrict__ hid) {
    extern __shared__ char smem[];
    unsigned short* z_sh   = (unsigned short*)smem;                   // 32*1792 bf16
    unsigned short* pre_sh = (unsigned short*)(smem + 114688);        // 32*1792 bf16
    unsigned short* h_sh   = (unsigned short*)(smem + 2 * 114688);    // 32*256 bf16

    const int tid = threadIdx.x;
    const int lane = tid & 31;
    const int wave = tid >> 5;                 // 0..31
    const int mt = wave >> 4;                  // each wave's 7 tiles share one Mtile
    const int KB = 256;                        // W_bot = Wt[n][256 + k]
    const unsigned pre_lds_off = (unsigned)(size_t)(void*)pre_sh;

    // gate-phase ownership: batch gb, features jb..jb+7
    const int gb = tid >> 5;
    const int jb = (tid & 31) * 8;

    for (int i = tid; i < 32 * 256; i += 1024) h_sh[i] = 0;
    float creg[8], cbreg[8];
#pragma unroll
    for (int i = 0; i < 8; i++) { creg[i] = 0.f; cbreg[i] = 0.f; }
    __syncthreads();

    const unsigned short* arow = h_sh + (size_t)(mt * 16 + (lane & 15)) * 256;
    const int mbase = mt * 16 + ((lane < 16) ? 0 : 8);

    for (int t = 0; t < 511; t++) {
#if HAVE_TDM
        // kick off async DMA of this step's pre slab while we do the GEMM
        if (wave == 0) tdm_load_pre(pre_bf + (size_t)t * 57344, pre_lds_off);
#endif
        // --- A fragments: h tile from LDS, loaded once per step ---
        v16bf a[8];
#pragma unroll
        for (int kf = 0; kf < 8; kf++) a[kf] = load_a_frag(arow, kf * 32, lane);

        // --- z = h @ W_bot : 7 Ntiles per wave ---
#pragma unroll 1
        for (int ti = 0; ti < 7; ti++) {
            const int tile = wave * 7 + ti;
            const int nt = tile % 112;
            const int n = nt * 16 + (lane & 15);
            if (ti < 6)   // prefetch next tile's Wt block (global_prefetch_b8)
                __builtin_prefetch(Wt + (size_t)(n + 16) * 512 + KB, 0, 1);
            v8f acc = {};
#pragma unroll
            for (int kf = 0; kf < 8; kf++) {
                v16bf bm = load_b_frag_t(Wt, KB + kf * 32, n, lane);
                acc = wmma_bf16(a[kf], bm, acc);
            }
#pragma unroll
            for (int r = 0; r < 8; r++)
                z_sh[(mbase + r) * 1792 + n] = f2bf(acc[r]);
        }
#if HAVE_TDM
        __builtin_amdgcn_s_wait_tensorcnt(0);   // pre_sh ready (no-op for waves 1..31)
#endif
        __syncthreads();

        // --- gates ---
#if HAVE_TDM
        const unsigned short* psrc = pre_sh;
#else
        const unsigned short* psrc = pre_bf + (size_t)t * 57344;
#endif
        const float dt = dtime[gb * 512 + (t + 1)];
        u16x8 pz[7], zz[7];
#pragma unroll
        for (int gg = 0; gg < 7; gg++) {
            const int off = gb * 1792 + gg * 256 + jb;   // 16B aligned
            pz[gg] = *(const u16x8*)(psrc + off);
            zz[gg] = *(const u16x8*)(z_sh + off);
        }
        u16x8 hv;
#pragma unroll
        for (int i = 0; i < 8; i++) {
            float g[7];
#pragma unroll
            for (int gg = 0; gg < 7; gg++) g[gg] = bf2f(pz[gg][i]) + bf2f(zz[gg][i]);
            const float gi = sigm(g[0]), gf = sigm(g[1]), go = sigm(g[2]);
            const float gpc = tanhf(g[3]);
            const float gib = sigm(g[4]), gfb = sigm(g[5]);
            const float gd = softplus(g[6]);
            const float cell = gf * creg[i] + gi * gpc;
            const float cbar = gfb * cbreg[i] + gib * gpc;
            const float cnext = cbar + (cell - cbar) * __expf(-gd * dt);
            const float h = go * tanhf(cnext);
            creg[i] = cnext; cbreg[i] = cbar;
            hv[i] = f2bf(h);
        }
        *(u16x8*)(h_sh + gb * 256 + jb) = hv;                          // ds_store_b128
        *(u16x8*)(hid + ((size_t)gb * 511 + t) * 256 + jb) = hv;       // b128 store
        __syncthreads();   // protects h_sh (next GEMM) and pre_sh (next TDM write)
    }
}

// ---------- kernel 4: intensities + log-likelihoods ----------
// One wave per (b,t); lane k<20 computes lambda_k = softplus(h . Wl[k]).
__global__ void loglik(const unsigned short* __restrict__ hid,
                       const float* __restrict__ Wl, const int* __restrict__ event,
                       float* __restrict__ out) {
    const int gw = (blockIdx.x * blockDim.x + threadIdx.x) >> 5;
    const int lane = threadIdx.x & 31;
    if (gw >= 32 * 511) return;
    const int b = gw / 511, t = gw % 511;
    const unsigned short* h = hid + ((size_t)b * 511 + t) * 256;

    float dot = 0.f;
    if (lane < 20) {
        const uint4* hp = (const uint4*)h;            // 8 bf16 per uint4
        const float* w = Wl + (size_t)lane * 256;
#pragma unroll 4
        for (int jc = 0; jc < 32; jc++) {
            const uint4 hvv = hp[jc];
            const unsigned q[4] = {hvv.x, hvv.y, hvv.z, hvv.w};
#pragma unroll
            for (int e = 0; e < 4; e++) {
                union { unsigned u; float f; } lo, hi2;
                lo.u  = q[e] << 16;
                hi2.u = q[e] & 0xFFFF0000u;
                dot += lo.f * w[jc * 8 + 2 * e] + hi2.f * w[jc * 8 + 2 * e + 1];
            }
        }
    }
    const float lam = (lane < 20) ? softplus(dot) : 0.f;

    float s = lam;
#pragma unroll
    for (int off = 16; off > 0; off >>= 1) s += __shfl_xor(s, off, 32);

    const int tgt = event[b * 512 + (t + 1)];
    const float mask = (tgt < 20) ? 1.f : 0.f;   // total_num == OBS_NUM == 20
    const int tsel = (tgt < 20) ? tgt : 0;
    const float lam_t = __shfl(lam, tsel, 32);
    const float llt = logf(lam_t + EPSF) * mask;
    const float lls = logf(s + EPSF) * mask;

    if (lane == 0) {
        const int idx = b * 511 + t;
        out[idx]             = llt;   // llt * mask_c
        out[16352 + idx]     = llt;   // llt * mask_o (identical masks)
        out[2 * 16352 + idx] = lls;   // lls_c
        out[3 * 16352 + idx] = lls;   // lls_o (identical: first 20 of 20)
    }
}

extern "C" void kernel_launch(void* const* d_in, const int* in_sizes, int n_in,
                              void* d_out, int out_size, void* d_ws, size_t ws_size,
                              hipStream_t stream) {
    (void)in_sizes; (void)n_in; (void)out_size; (void)ws_size;
    const int*   event = (const int*)d_in[0];
    const float* dtime = (const float*)d_in[1];
    const float* Emb   = (const float*)d_in[2];
    const float* W     = (const float*)d_in[3];
    const float* bias  = (const float*)d_in[4];
    const float* Wl    = (const float*)d_in[5];
    float* out = (float*)d_out;

    char* ws = (char*)d_ws;
    unsigned short* Wt     = (unsigned short*)ws;                     // 1792*512 bf16
    unsigned short* Embbf  = (unsigned short*)(ws + (size_t)512 * 1792 * 2);
    unsigned short* pre_bf =
        (unsigned short*)(ws + (size_t)512 * 1792 * 2 + (size_t)23 * 256 * 2);
    unsigned short* hid =
        (unsigned short*)((char*)pre_bf + (size_t)16352 * 1792 * 2);  // 16352*256 bf16

    convert_bf16<<<(512 * 1792 + 255) / 256, 256, 0, stream>>>(W, Emb, Wt, Embbf);

    // 1022 Mtiles x 28 Ngroups = 28616 waves; 8 waves / block -> 3577 blocks
    pre_gemm<<<3577, 256, 0, stream>>>(event, Embbf, Wt, bias, pre_bf);

    const size_t lds_bytes = 2 * 114688 + 32 * 256 * 2;               // 240KB
    recurrent_scan<<<1, 1024, lds_bytes, stream>>>(dtime, Wt, pre_bf, hid);

    // 16352 waves -> 2044 blocks of 256 threads
    loglik<<<2044, 256, 0, stream>>>(hid, Wl, event, out);
}
